// GCN_3616362463494
// MI455X (gfx1250) — matrix operand order repaired and checked by reading, not compiled, not run on hardware
//
#include <hip/hip_runtime.h>
#include <hip/hip_bf16.h>

typedef __attribute__((ext_vector_type(16))) _Float16 v16h;
typedef __attribute__((ext_vector_type(8)))  float    v8f;

#define N_NODES 100000
#define N_EDGES 1600000
#define K_IN    100
#define C_HID   32
#define K_PAD   128
#define M_TILES (N_NODES / 16)   // 6250, exact

// ---------------- kernel: pack W_conv [100x32] f32 -> Wt [32][128] f16 (zero-padded K)
__global__ void k_pack_w(const float* __restrict__ W, _Float16* __restrict__ Wt) {
  int t = blockIdx.x * blockDim.x + threadIdx.x;
  if (t >= C_HID * K_PAD) return;
  int n = t >> 7;          // output channel 0..31
  int k = t & (K_PAD - 1); // 0..127
  float v = (k < K_IN) ? W[k * C_HID + n] : 0.0f;
  Wt[n * K_PAD + k] = (_Float16)v;
}

// ---------------- kernel: deg = 1 (self-loop)
__global__ void k_init_deg(float* __restrict__ deg) {
  int i = blockIdx.x * blockDim.x + threadIdx.x;
  if (i < N_NODES) deg[i] = 1.0f;
}

// ---------------- kernel: deg[dst] += 1 per edge
__global__ void k_edge_deg(const int* __restrict__ ei, float* __restrict__ deg) {
  int e = blockIdx.x * blockDim.x + threadIdx.x;
  if (e < N_EDGES) atomicAdd(&deg[ei[N_EDGES + e]], 1.0f);
}

// ---------------- kernel: dinv = rsqrt(deg) in place
__global__ void k_rsqrt(float* __restrict__ deg) {
  int i = blockIdx.x * blockDim.x + threadIdx.x;
  if (i < N_NODES) deg[i] = rsqrtf(fmaxf(deg[i], 1e-12f));
}

// ---------------- kernel: h = x @ W_conv via WMMA f32<-f16, one wave per 16x32 tile
__global__ void __launch_bounds__(256) k_gemm_wmma(const float* __restrict__ x,
                                                   const _Float16* __restrict__ Wt,
                                                   float* __restrict__ h) {
  const int wave = threadIdx.x >> 5;
  const int lane = threadIdx.x & 31;
  const int tile = blockIdx.x * 8 + wave;
  if (tile >= M_TILES) return;            // uniform per wave; EXEC all-1 at WMMAs

  const int half = lane >> 4;             // lane-group select (ISA A/B layout)
  const int nloc = lane & 15;             // column within 16-wide tile / row for A
  const int row  = tile * 16 + nloc;      // A-matrix row this lane feeds
  const float* xr = x + (size_t)row * K_IN;

  v8f c0 = {};
  v8f c1 = {};

  #pragma unroll
  for (int s = 0; s < 4; ++s) {
    const int k0 = s * 32;
    // A fragment: contiguous K order, same as B. element j   -> K = k0 + half*16 + j
    //                                    element j+8 -> K = k0 + half*16 + 8 + j
    v16h a;
    #pragma unroll
    for (int j = 0; j < 8; ++j) {
      int k1 = k0 + half * 16 + j;
      int k2 = k1 + 8;
      a[j]     = (_Float16)((k1 < K_IN) ? xr[k1] : 0.0f);
      a[j + 8] = (_Float16)((k2 < K_IN) ? xr[k2] : 0.0f);
    }
    // B fragment: column n = nloc(+16), element t -> K = k0 + half*16 + t
    const _Float16* bp0 = Wt + (size_t)(nloc)      * K_PAD + k0 + half * 16;
    const _Float16* bp1 = Wt + (size_t)(16 + nloc) * K_PAD + k0 + half * 16;
    v16h b0, b1;
    #pragma unroll
    for (int t = 0; t < 16; ++t) { b0[t] = bp0[t]; b1[t] = bp1[t]; }

    c0 = __builtin_amdgcn_wmma_f32_16x16x32_f16(false, a, false, b0,
                                                (short)0, c0, false, false);
    c1 = __builtin_amdgcn_wmma_f32_16x16x32_f16(false, a, false, b1,
                                                (short)0, c1, false, false);
  }

  // C/D f32 layout: VGPR i -> M = i + 8*half, N = lane&15
  #pragma unroll
  for (int i = 0; i < 8; ++i) {
    int m = tile * 16 + i + 8 * half;
    h[(size_t)m * C_HID + nloc]      = c0[i];
    h[(size_t)m * C_HID + 16 + nloc] = c1[i];
  }
}

// ---------------- kernel: agg = h * dinv^2  (self-loop message, also initializes agg)
__global__ void k_selfloop(const float* __restrict__ h, const float* __restrict__ dinv,
                           float* __restrict__ agg) {
  int t = blockIdx.x * blockDim.x + threadIdx.x;
  if (t >= N_NODES * C_HID) return;
  int node = t >> 5;
  float di = dinv[node];
  agg[t] = h[t] * di * di;
}

// ---------------- kernel: edge messages, 8 lanes per edge, float4 chunks
__global__ void k_edge_msg(const int* __restrict__ ei, const float* __restrict__ dinv,
                           const float* __restrict__ h, float* __restrict__ agg) {
  int t = blockIdx.x * blockDim.x + threadIdx.x;
  if (t >= N_EDGES * 8) return;
  int e   = t >> 3;
  int sub = t & 7;
  int src = ei[e];
  int dst = ei[N_EDGES + e];
  float norm = dinv[src] * dinv[dst];
  const float4* hv = (const float4*)h;
  float4 v = hv[src * 8 + sub];
  float* ap = agg + dst * C_HID + sub * 4;
  atomicAdd(ap + 0, v.x * norm);
  atomicAdd(ap + 1, v.y * norm);
  atomicAdd(ap + 2, v.z * norm);
  atomicAdd(ap + 3, v.w * norm);
}

// ---------------- kernel: bias+relu, 3-layer MLP, log_softmax (weights in LDS)
__global__ void __launch_bounds__(256) k_mlp(const float* __restrict__ agg,
    const float* __restrict__ bc,
    const float* __restrict__ W1, const float* __restrict__ b1,
    const float* __restrict__ W2, const float* __restrict__ b2,
    const float* __restrict__ W3, const float* __restrict__ b3,
    float* __restrict__ out) {
  __shared__ float sW1[32 * 16], sW2[16 * 8], sW3[8 * 10];
  __shared__ float sbc[32], sb1[16], sb2[8], sb3[10];
  for (int t = threadIdx.x; t < 512; t += 256) sW1[t] = W1[t];
  for (int t = threadIdx.x; t < 128; t += 256) sW2[t] = W2[t];
  for (int t = threadIdx.x; t < 80;  t += 256) sW3[t] = W3[t];
  if (threadIdx.x < 32) sbc[threadIdx.x] = bc[threadIdx.x];
  if (threadIdx.x < 16) sb1[threadIdx.x] = b1[threadIdx.x];
  if (threadIdx.x < 8)  sb2[threadIdx.x] = b2[threadIdx.x];
  if (threadIdx.x < 10) sb3[threadIdx.x] = b3[threadIdx.x];
  __syncthreads();

  int node = blockIdx.x * blockDim.x + threadIdx.x;
  if (node >= N_NODES) return;

  float a0[32];
  #pragma unroll
  for (int j = 0; j < 32; ++j) {
    float v = agg[node * C_HID + j] + sbc[j];
    a0[j] = v > 0.0f ? v : 0.0f;
  }
  float a1[16];
  #pragma unroll
  for (int o = 0; o < 16; ++o) {
    float s = sb1[o];
    #pragma unroll
    for (int j = 0; j < 32; ++j) s = fmaf(a0[j], sW1[j * 16 + o], s);
    a1[o] = s > 0.0f ? s : 0.0f;
  }
  float a2[8];
  #pragma unroll
  for (int o = 0; o < 8; ++o) {
    float s = sb2[o];
    #pragma unroll
    for (int j = 0; j < 16; ++j) s = fmaf(a1[j], sW2[j * 8 + o], s);
    a2[o] = s > 0.0f ? s : 0.0f;
  }
  float z[10];
  #pragma unroll
  for (int o = 0; o < 10; ++o) {
    float s = sb3[o];
    #pragma unroll
    for (int j = 0; j < 8; ++j) s = fmaf(a2[j], sW3[j * 10 + o], s);
    z[o] = s;
  }
  float m = z[0];
  #pragma unroll
  for (int o = 1; o < 10; ++o) m = fmaxf(m, z[o]);
  float s = 0.0f;
  #pragma unroll
  for (int o = 0; o < 10; ++o) s += expf(z[o] - m);
  float l = logf(s);
  #pragma unroll
  for (int o = 0; o < 10; ++o) out[node * 10 + o] = z[o] - m - l;
}

extern "C" void kernel_launch(void* const* d_in, const int* in_sizes, int n_in,
                              void* d_out, int out_size, void* d_ws, size_t ws_size,
                              hipStream_t stream) {
  const float* x  = (const float*)d_in[0];
  const int*   ei = (const int*)d_in[1];   // [2, E] int32 (jax x64 disabled)
  // d_in[2] edge_weight: unused by the reference forward
  const float* Wc = (const float*)d_in[3];
  const float* bc = (const float*)d_in[4];
  const float* W1 = (const float*)d_in[5];
  const float* b1 = (const float*)d_in[6];
  const float* W2 = (const float*)d_in[7];
  const float* b2 = (const float*)d_in[8];
  const float* W3 = (const float*)d_in[9];
  const float* b3 = (const float*)d_in[10];
  float* out = (float*)d_out;

  // workspace layout (all offsets 32B-aligned)
  char* ws = (char*)d_ws;
  float*    h    = (float*)(ws);                       // 100000*32 f32 = 12.8 MB
  float*    agg  = (float*)(ws + 12800000);            // 100000*32 f32 = 12.8 MB
  float*    dinv = (float*)(ws + 25600000);            // 100000 f32 = 0.4 MB
  _Float16* Wt   = (_Float16*)(ws + 26000000);         // 32*128 f16 = 8 KB

  k_pack_w   <<<(C_HID * K_PAD + 255) / 256, 256, 0, stream>>>(Wc, Wt);
  k_init_deg <<<(N_NODES + 255) / 256,       256, 0, stream>>>(dinv);
  k_edge_deg <<<(N_EDGES + 255) / 256,       256, 0, stream>>>(ei, dinv);
  k_rsqrt    <<<(N_NODES + 255) / 256,       256, 0, stream>>>(dinv);
  k_gemm_wmma<<<(M_TILES + 7) / 8,           256, 0, stream>>>(x, Wt, h);
  k_selfloop <<<(N_NODES * C_HID + 255)/256, 256, 0, stream>>>(h, dinv, agg);
  k_edge_msg <<<(N_EDGES * 8 + 255) / 256,   256, 0, stream>>>(ei, dinv, h, agg);
  k_mlp      <<<(N_NODES + 255) / 256,       256, 0, stream>>>(agg, bc, W1, b1, W2, b2, W3, b3, out);
}
